// CausalSelfAttention_35579509080713
// MI455X (gfx1250) — matrix operand order repaired and checked by reading, not compiled, not run on hardware
//
#include <hip/hip_runtime.h>

typedef __bf16 bf16_t;
typedef __attribute__((ext_vector_type(16))) __bf16 v16bf;
typedef __attribute__((ext_vector_type(8)))  float  v8f;
typedef int v4i_vs __attribute__((vector_size(16)));   // pointee type for async-LDS builtin

// ---------------------------------------------------------------------------
// Async global->LDS staging (CDNA5 GLOBAL_LOAD_ASYNC_TO_LDS_B128, ASYNCcnt).
// Signature (from hipcc diagnostic): (AS1 v4i*, AS3 v4i*, imm offset, imm cpol)
// Guarded: falls back to a synchronous 16B copy if the builtin is absent.
// ---------------------------------------------------------------------------
#if __has_builtin(__builtin_amdgcn_global_load_async_to_lds_b128)
#define USE_ASYNC_LDS 1
#else
#define USE_ASYNC_LDS 0
#endif

__device__ __forceinline__ void stage16(bf16_t* l, const bf16_t* g) {
#if USE_ASYNC_LDS
  __builtin_amdgcn_global_load_async_to_lds_b128(
      (__attribute__((address_space(1))) v4i_vs*)g,
      (__attribute__((address_space(3))) v4i_vs*)l, 0, 0);
#else
  *(uint4*)l = *(const uint4*)g;
#endif
}
__device__ __forceinline__ void wait_async0() {
#if USE_ASYNC_LDS
#if __has_builtin(__builtin_amdgcn_s_wait_asynccnt)
  __builtin_amdgcn_s_wait_asynccnt(0);
#else
  asm volatile("s_wait_asynccnt 0" ::: "memory");
#endif
#endif
}

// ---------------------------------------------------------------------------
// Fragment helpers (wave32 WMMA layouts per CDNA5 ISA §7.12.2)
// A 16x32 bf16: lane holds row (lane&15); K groups at half*8 and half*8+16.
// B 32x16 bf16: lane holds col (lane&15); K groups at half*8 and half*8+16.
// ---------------------------------------------------------------------------
__device__ __forceinline__ v16bf load_frag_a(const bf16_t* p) {
  union { v16bf v; uint4 q[2]; } u;
  const uint4* p4 = (const uint4*)p;
  u.q[0] = p4[0];
  u.q[1] = p4[2];
  return u.v;
}
__device__ __forceinline__ v16bf load_frag_b(const bf16_t* p) {
  union { v16bf v; uint4 q[2]; } u;
  const uint4* p4 = (const uint4*)p;
  u.q[0] = p4[0];
  u.q[1] = p4[2];
  return u.v;
}
__device__ __forceinline__ v8f wmma_bf16(v16bf a, v16bf b, v8f c) {
  return __builtin_amdgcn_wmma_f32_16x16x32_bf16(
      /*neg_a=*/false, a, /*neg_b=*/false, b,
      /*c_mod=*/(short)0, c, /*reuse_a=*/false, /*reuse_b=*/false);
}

// ---------------------------------------------------------------------------
// Problem constants
// ---------------------------------------------------------------------------
#define BB   4
#define SS   2048
#define EE   1024
#define HH   16
#define DD   64
#define MM   (BB * SS)        // 8192 tokens

// ---------------------------------------------------------------------------
// Stage 0a: fp32 -> bf16 conversion (x)
// ---------------------------------------------------------------------------
__global__ __launch_bounds__(256) void cvt_kernel(const float* __restrict__ X,
                                                  bf16_t* __restrict__ Xb,
                                                  int n8) {
  int i = blockIdx.x * 256 + threadIdx.x;
  if (i >= n8) return;
  const float4* p = (const float4*)(X + (size_t)i * 8);
  float4 a = p[0], b = p[1];
  union { bf16_t h[8]; uint4 q; } u;
  u.h[0] = (bf16_t)a.x; u.h[1] = (bf16_t)a.y;
  u.h[2] = (bf16_t)a.z; u.h[3] = (bf16_t)a.w;
  u.h[4] = (bf16_t)b.x; u.h[5] = (bf16_t)b.y;
  u.h[6] = (bf16_t)b.z; u.h[7] = (bf16_t)b.w;
  ((uint4*)Xb)[i] = u.q;
}

// ---------------------------------------------------------------------------
// Stage 0b: W [K,N] fp32 -> Wt [N,K] bf16 (LDS tiled transpose)
// ---------------------------------------------------------------------------
__global__ __launch_bounds__(256) void transpose_kernel(
    const float* __restrict__ W, bf16_t* __restrict__ Wt, int Kd, int Nd) {
  __shared__ float tile[32][33];
  int tx = threadIdx.x & 31, ty = threadIdx.x >> 5;   // ty: 0..7
  int nb = blockIdx.x * 32, kb = blockIdx.y * 32;
  #pragma unroll
  for (int i = 0; i < 32; i += 8)
    tile[ty + i][tx] = W[(size_t)(kb + ty + i) * Nd + nb + tx];
  __syncthreads();
  #pragma unroll
  for (int i = 0; i < 32; i += 8)
    Wt[(size_t)(nb + ty + i) * Kd + kb + tx] = (bf16_t)tile[tx][ty + i];
}

// ---------------------------------------------------------------------------
// Tiled WMMA GEMM:  C[M,N] = A[M,K] * Bt[N,K]^T + bias
//   workgroup: 256 thr (8 waves), tile 128x128, K-step 32, DOUBLE-BUFFERED:
//   async-stage tile k+1 into buf^1 while 8 WMMAs run out of buf.
//   wave grid 4(m) x 2(n): each wave 32x64 = 2x4 accumulator tiles
// mode 0: plain fp32 row-major output (proj)
// mode 1: split QKV epilogue -> Q[B,H,S,D]*qscale, K[B,H,S,D], Vt[B,H,D,S]
// ---------------------------------------------------------------------------
#define GLD 40   // LDS row stride (bf16 elems), padded
__global__ __launch_bounds__(256) void gemm_kernel(
    const bf16_t* __restrict__ A, const bf16_t* __restrict__ Bt,
    const float* __restrict__ bias, int N, int mode, float qscale,
    bf16_t* __restrict__ Qo, bf16_t* __restrict__ Ko,
    bf16_t* __restrict__ Vt, float* __restrict__ Co) {
  __shared__ bf16_t sA[2][128 * GLD];
  __shared__ bf16_t sB[2][128 * GLD];
  const int tid  = threadIdx.x;
  const int lane = tid & 31;
  const int wid  = tid >> 5;
  const int ln = lane & 15, hf = lane >> 4;
  const int wm = wid & 3, wn = wid >> 2;
  const int m0 = blockIdx.y * 128;
  const int n0 = blockIdx.x * 128;
  const int K  = EE;

  // this thread's two 16B staging chunks (same for every K-step)
  const int idx0 = tid * 2;
  const int row0 = idx0 >> 2, off0 = (idx0 & 3) * 8;
  const int row1 = (idx0 + 1) >> 2, off1 = ((idx0 + 1) & 3) * 8;
  const bf16_t* gA0 = &A [(size_t)(m0 + row0) * K + off0];
  const bf16_t* gA1 = &A [(size_t)(m0 + row1) * K + off1];
  const bf16_t* gB0 = &Bt[(size_t)(n0 + row0) * K + off0];
  const bf16_t* gB1 = &Bt[(size_t)(n0 + row1) * K + off1];
  const int lA0 = row0 * GLD + off0, lA1 = row1 * GLD + off1;

  v8f acc[2][4];
  #pragma unroll
  for (int i = 0; i < 2; ++i)
    #pragma unroll
    for (int j = 0; j < 4; ++j) acc[i][j] = {};

  // prologue: stage k0 = 0 into buffer 0
  stage16(&sA[0][lA0], gA0);
  stage16(&sA[0][lA1], gA1);
  stage16(&sB[0][lA0], gB0);
  stage16(&sB[0][lA1], gB1);
  wait_async0();
  __syncthreads();

  for (int k0 = 0; k0 < K; k0 += 32) {
    const int buf = (k0 >> 5) & 1;
    if (k0 + 32 < K) {                       // prefetch next tile under compute
      stage16(&sA[buf ^ 1][lA0], gA0 + k0 + 32);
      stage16(&sA[buf ^ 1][lA1], gA1 + k0 + 32);
      stage16(&sB[buf ^ 1][lA0], gB0 + k0 + 32);
      stage16(&sB[buf ^ 1][lA1], gB1 + k0 + 32);
    }
    v16bf aF[2], bF[4];
    #pragma unroll
    for (int i = 0; i < 2; ++i)
      aF[i] = load_frag_a(&sA[buf][(wm * 32 + i * 16 + ln) * GLD + hf * 8]);
    #pragma unroll
    for (int j = 0; j < 4; ++j)
      bF[j] = load_frag_b(&sB[buf][(wn * 64 + j * 16 + ln) * GLD + hf * 8]);
    #pragma unroll
    for (int i = 0; i < 2; ++i)
      #pragma unroll
      for (int j = 0; j < 4; ++j)
        acc[i][j] = wmma_bf16(aF[i], bF[j], acc[i][j]);
    wait_async0();
    __syncthreads();
  }

  // Epilogue. C tile layout: VGPR r -> row (r + hf*8), lane ln -> col.
  #pragma unroll
  for (int i = 0; i < 2; ++i) {
    int mbase = m0 + wm * 32 + i * 16 + hf * 8;     // token (aligned to 8)
    int bb = mbase >> 11;                           // / 2048
    int sidx = mbase & 2047;
    #pragma unroll
    for (int j = 0; j < 4; ++j) {
      int ng = n0 + wn * 64 + j * 16 + ln;
      float bv = bias[ng];
      v8f c = acc[i][j];
      if (mode == 0) {
        #pragma unroll
        for (int r = 0; r < 8; ++r)
          Co[(size_t)(mbase + r) * N + ng] = c[r] + bv;
      } else {
        int region = ng >> 10;                      // uniform per workgroup
        int f = ng & 1023;
        int hh = f >> 6, d = f & 63;
        if (region == 0) {                          // Q (scaled)
          size_t base = ((size_t)(bb * HH + hh) * SS + sidx) * DD + d;
          #pragma unroll
          for (int r = 0; r < 8; ++r)
            Qo[base + (size_t)r * DD] = (bf16_t)((c[r] + bv) * qscale);
        } else if (region == 1) {                   // K
          size_t base = ((size_t)(bb * HH + hh) * SS + sidx) * DD + d;
          #pragma unroll
          for (int r = 0; r < 8; ++r)
            Ko[base + (size_t)r * DD] = (bf16_t)(c[r] + bv);
        } else {                                    // V transposed [b,h,d,s]
          size_t base = ((size_t)(bb * HH + hh) * DD + d) * SS + sidx;
          union { bf16_t h[8]; uint4 q; } u;
          #pragma unroll
          for (int r = 0; r < 8; ++r) u.h[r] = (bf16_t)(c[r] + bv);
          *(uint4*)&Vt[base] = u.q;                 // 8 contiguous s -> b128
        }
      }
    }
  }
}

// ---------------------------------------------------------------------------
// Stage 2: causal flash attention.
//   grid (S/64, B*H), 128 thr = 4 waves; wave w owns q rows [qt+16w, +16).
//   Q pre-scaled by D^-0.5 * log2(e) -> softmax in base-2 (v_exp_f32).
//   K tiles of 64 keys async-staged in LDS; Vt [d][s] staged in LDS.
// ---------------------------------------------------------------------------
#define ALD 72   // LDS row stride (bf16 elems)
__global__ __launch_bounds__(128) void attn_kernel(
    const bf16_t* __restrict__ Q, const bf16_t* __restrict__ Kc,
    const bf16_t* __restrict__ Vt, bf16_t* __restrict__ Y) {
  __shared__ bf16_t sK[64 * ALD];
  __shared__ bf16_t sV[64 * ALD];
  __shared__ bf16_t sP[4][16 * ALD];
  const int tid  = threadIdx.x;
  const int lane = tid & 31;
  const int wid  = tid >> 5;
  const int ln = lane & 15, hf = lane >> 4;
  const int bh = blockIdx.y;            // b*16 + h
  const int b  = bh >> 4, h = bh & 15;
  const int qt = blockIdx.x * 64;
  const int q0 = qt + wid * 16;
  const size_t headOff = (size_t)bh * SS * DD;   // == bh * DD * SS for Vt too
  const bf16_t* Qp = Q  + headOff;
  const bf16_t* Kp = Kc + headOff;
  const bf16_t* Vp = Vt + headOff;

  // Q fragments held in registers for the whole kernel (A-frag layout).
  v16bf qf[2];
  #pragma unroll
  for (int t = 0; t < 2; ++t) {
    const uint4* p = (const uint4*)&Qp[(size_t)(q0 + ln) * DD + t * 32 + hf * 8];
    union { v16bf v; uint4 q[2]; } u;
    u.q[0] = p[0]; u.q[1] = p[2];
    qf[t] = u.v;
  }

  v8f o[4];
  float mst[8], lst[8];
  #pragma unroll
  for (int t = 0; t < 4; ++t) o[t] = {};
  #pragma unroll
  for (int r = 0; r < 8; ++r) { mst[r] = -1e30f; lst[r] = 0.f; }

  for (int kb = 0; kb <= qt; kb += 64) {
    __syncthreads();                              // tiles free to overwrite
    #pragma unroll
    for (int c0 = 0; c0 < 4; ++c0) {              // 2x 8KB tiles, 512 chunks
      int c = tid + c0 * 128;
      int row = c >> 3, off = (c & 7) * 8;
      stage16(&sK[row * ALD + off], &Kp[(size_t)(kb + row) * DD + off]);
      stage16(&sV[row * ALD + off], &Vp[(size_t)row * SS + kb + off]);
    }
    wait_async0();
    __syncthreads();

    // scores: S = Q * K^T over D=64 (2 WMMA per 16-key tile)
    v8f st[4];
    #pragma unroll
    for (int j = 0; j < 4; ++j) {
      v16bf k0 = load_frag_b(&sK[(j * 16 + ln) * ALD + hf * 8]);       // d 0..31
      v16bf k1 = load_frag_b(&sK[(j * 16 + ln) * ALD + 32 + hf * 8]);  // d 32..63
      v8f s = {};
      s = wmma_bf16(qf[0], k0, s);
      s = wmma_bf16(qf[1], k1, s);
      st[j] = s;
    }

    if (kb == qt) {                                  // diagonal tile: causal mask
      #pragma unroll
      for (int j = 0; j < 4; ++j)
        #pragma unroll
        for (int r = 0; r < 8; ++r) {
          int m = q0 + hf * 8 + r;
          int n = kb + j * 16 + ln;
          if (n > m) st[j][r] = -1e30f;
        }
    }

    // online softmax (base-2 domain; row spans 16 lanes of a half-wave)
    #pragma unroll
    for (int r = 0; r < 8; ++r) {
      float mx = fmaxf(fmaxf(st[0][r], st[1][r]), fmaxf(st[2][r], st[3][r]));
      mx = fmaxf(mx, __shfl_xor(mx, 1, 16));
      mx = fmaxf(mx, __shfl_xor(mx, 2, 16));
      mx = fmaxf(mx, __shfl_xor(mx, 4, 16));
      mx = fmaxf(mx, __shfl_xor(mx, 8, 16));
      float mnew  = fmaxf(mst[r], mx);
      float alpha = __builtin_exp2f(mst[r] - mnew);
      mst[r] = mnew;
      float rs = 0.f;
      #pragma unroll
      for (int j = 0; j < 4; ++j) {
        float p = __builtin_exp2f(st[j][r] - mnew);
        st[j][r] = p;
        rs += p;
      }
      rs += __shfl_xor(rs, 1, 16);
      rs += __shfl_xor(rs, 2, 16);
      rs += __shfl_xor(rs, 4, 16);
      rs += __shfl_xor(rs, 8, 16);
      lst[r] = lst[r] * alpha + rs;
      #pragma unroll
      for (int t = 0; t < 4; ++t) o[t][r] *= alpha;
    }

    // P (C-layout) -> per-wave LDS -> A-fragment layout.  Same-wave DS ops
    // are in-order (DScnt), no workgroup barrier needed for this buffer.
    bf16_t* Pb = &sP[wid][0];
    #pragma unroll
    for (int j = 0; j < 4; ++j)
      #pragma unroll
      for (int r = 0; r < 8; ++r)
        Pb[(hf * 8 + r) * ALD + j * 16 + ln] = (bf16_t)st[j][r];

    v16bf pf0 = load_frag_a(&Pb[ln * ALD + hf * 8]);        // k 0..31
    v16bf pf1 = load_frag_a(&Pb[ln * ALD + 32 + hf * 8]);   // k 32..63

    // O += P * V  (V^T rows are d -> clean B-fragments)
    #pragma unroll
    for (int t = 0; t < 4; ++t) {
      v16bf v0 = load_frag_b(&sV[(t * 16 + ln) * ALD + hf * 8]);
      v16bf v1 = load_frag_b(&sV[(t * 16 + ln) * ALD + 32 + hf * 8]);
      o[t] = wmma_bf16(pf0, v0, o[t]);
      o[t] = wmma_bf16(pf1, v1, o[t]);
    }
  }

  // normalize and write Y [B*S, E] as bf16 for the projection GEMM
  #pragma unroll
  for (int t = 0; t < 4; ++t)
    #pragma unroll
    for (int r = 0; r < 8; ++r) {
      int s   = q0 + hf * 8 + r;
      int col = h * DD + t * 16 + ln;
      float val = o[t][r] / lst[r];
      Y[((size_t)(b * SS + s)) * EE + col] = (bf16_t)val;
    }
}

// ---------------------------------------------------------------------------
// Host launch
// ---------------------------------------------------------------------------
extern "C" void kernel_launch(void* const* d_in, const int* in_sizes, int n_in,
                              void* d_out, int out_size, void* d_ws, size_t ws_size,
                              hipStream_t stream) {
  const float* x      = (const float*)d_in[0];   // [B,S,E]
  const float* w_qkv  = (const float*)d_in[1];   // [E,3E]
  const float* b_qkv  = (const float*)d_in[2];   // [3E]
  const float* w_proj = (const float*)d_in[3];   // [E,E]
  const float* b_proj = (const float*)d_in[4];   // [E]
  float* out = (float*)d_out;

  char* ws = (char*)d_ws;
  const size_t MB = 1024 * 1024;
  bf16_t* xb     = (bf16_t*)(ws);            // 16 MB  [M,E]
  bf16_t* wqkvT  = (bf16_t*)(ws + 16 * MB);  //  6 MB  [3E,E]
  bf16_t* wprojT = (bf16_t*)(ws + 22 * MB);  //  2 MB  [E,E]
  bf16_t* Qb     = (bf16_t*)(ws + 24 * MB);  // 16 MB  [B,H,S,D]
  bf16_t* Kb     = (bf16_t*)(ws + 40 * MB);  // 16 MB  [B,H,S,D]
  bf16_t* Vtb    = (bf16_t*)(ws + 56 * MB);  // 16 MB  [B,H,D,S]
  bf16_t* Yb     = (bf16_t*)(ws + 72 * MB);  // 16 MB  [M,E]

  // fold head scaling and log2(e) into Q so softmax runs on v_exp (base 2)
  const float qscale = 0.125f * 1.44269504088896340736f;  // D^-0.5 * log2(e)

  cvt_kernel<<<(MM * EE / 8 + 255) / 256, 256, 0, stream>>>(x, xb, MM * EE / 8);
  transpose_kernel<<<dim3(3 * EE / 32, EE / 32), 256, 0, stream>>>(
      w_qkv, wqkvT, EE, 3 * EE);
  transpose_kernel<<<dim3(EE / 32, EE / 32), 256, 0, stream>>>(
      w_proj, wprojT, EE, EE);

  gemm_kernel<<<dim3(3 * EE / 128, MM / 128), 256, 0, stream>>>(
      xb, wqkvT, b_qkv, 3 * EE, /*mode=*/1, qscale, Qb, Kb, Vtb, nullptr);

  attn_kernel<<<dim3(SS / 64, BB * HH), 128, 0, stream>>>(Qb, Kb, Vtb, Yb);

  gemm_kernel<<<dim3(EE / 128, MM / 128), 256, 0, stream>>>(
      Yb, wprojT, b_proj, EE, /*mode=*/0, 1.0f, nullptr, nullptr, nullptr, out);
}
